// KnowledgeGraph_45552423142099
// MI455X (gfx1250) — compile-verified, matrix operand order checked by reading, not run on hardware
//
#include <hip/hip_runtime.h>
#include <hip/hip_bf16.h>
#include <stdint.h>

#define GAT_ALPHA 0.2f
#define GAT_NEG_INF -9.0e15f

typedef __attribute__((ext_vector_type(16))) __bf16 v16bf;
typedef __attribute__((ext_vector_type(8)))  float  v8f;

// Problem constants (from reference setup_inputs)
static constexpr int Bb = 16;
static constexpr int Nn = 1024;
static constexpr int Tt = 6;
static constexpr int Dd = 128;   // embedding dim
static constexpr int Hh = 4;     // heads
static constexpr int Lc = 400;   // CONV_MAX

// CDNA5 16-bit WMMA fragment K-mapping (A 16x32 / B 32x16 bf16):
// element e (0..15) of v16bf, lanehi = 8 if lane>=16 else 0  -> K in [0,32)
__device__ __forceinline__ int kmap(int e, int lanehi) {
  return e + ((e < 8) ? 0 : 8) + lanehi;
}

__device__ __forceinline__ float fsigmoid(float x) {
  return 1.0f / (1.0f + __expf(-x));
}

// ---------------------------------------------------------------------------
// Stage 0a: x[b,n,:] = sum_t emb[story[b,n,t]] + (valid ? dh[b, n-(kb-1)] : 0)
// ---------------------------------------------------------------------------
__global__ __launch_bounds__(128)
void build_x(const int* __restrict__ story, const int* __restrict__ kb_len,
             const int* __restrict__ conv_len, const float* __restrict__ dh,
             const float* __restrict__ emb, float* __restrict__ x)
{
  int n = blockIdx.x % Nn;
  int b = blockIdx.x / Nn;
  int d = threadIdx.x;
  const int* st = story + ((size_t)b * Nn + n) * Tt;
  float s = 0.0f;
#pragma unroll
  for (int t = 0; t < Tt; ++t)
    s += emb[(size_t)st[t] * Dd + d];
  int off = n - (kb_len[b] - 1);
  if (off >= 0 && off < conv_len[b])
    s += dh[((size_t)b * Lc + off) * Dd + d];
  x[((size_t)b * Nn + n) * Dd + d] = s;
}

// ---------------------------------------------------------------------------
// Stage 0b: adjacency (+diag rule) -> bitmask words  [B][N][N/32]
// ---------------------------------------------------------------------------
__global__ __launch_bounds__(32)
void build_mask(const float* __restrict__ adj, const int* __restrict__ kb_len,
                const int* __restrict__ conv_len, uint32_t* __restrict__ maskw)
{
  int i = blockIdx.x % Nn;
  int b = blockIdx.x / Nn;
  int start = kb_len[b] + conv_len[b];
  const float* arow = adj + ((size_t)b * Nn + i) * Nn;
  int w = threadIdx.x;            // word index 0..31 (N/32 == 32)
  uint32_t word = 0;
  int jbase = w * 32;
#pragma unroll
  for (int k = 0; k < 32; ++k) {
    int j = jbase + k;
    bool bit = arow[j] > 0.0f;
    if (j == i && i >= start) bit = true;
    word |= (bit ? 1u : 0u) << k;
  }
  maskw[((size_t)b * Nn + i) * (Nn / 32) + w] = word;
}

// ---------------------------------------------------------------------------
// Weight pre-transpose + bf16 convert: W[h,k,f] (f32) -> WT[h,f,k] (bf16)
// ---------------------------------------------------------------------------
__global__ __launch_bounds__(256)
void conv_w(const float* __restrict__ W, __bf16* __restrict__ WT, int DIN)
{
  int idx = blockIdx.x * 256 + threadIdx.x;       // idx = (h*64+f)*DIN + k
  int total = Hh * DIN * 64;
  if (idx >= total) return;
  int k = idx % DIN;
  int rest = idx / DIN;
  int f = rest % 64;
  int h = rest / 64;
  WT[idx] = (__bf16)W[((size_t)h * DIN + k) * 64 + f];
}

// ---------------------------------------------------------------------------
// Per-layer projection: WhT[b,h,f,n] = (x[b,n,:] @ W[h,:,:])^T  (bf16,
// f-major so the attention kernel's B-fragments are K-contiguous),
// plus f1/f2 = Wh . a[:, :F] / a[:, F:]. One wave per 16-row tile.
// Loads issued first, VALU conversion in their shadow, then WMMAs.
// ---------------------------------------------------------------------------
template<int DIN>
__global__ __launch_bounds__(32)
void gat_wh(const float* __restrict__ xin,   // [B,N,DIN] f32
            const __bf16* __restrict__ WT,   // [H,64,DIN] bf16 (transposed)
            const float* __restrict__ avec,  // [H,128]
            __bf16* __restrict__ WhT,        // [B,H,64,N] bf16
            float* __restrict__ f1, float* __restrict__ f2)
{
  const int ntiles = Nn / 16;
  int it = blockIdx.x % ntiles;
  int bh = blockIdx.x / ntiles;
  int h = bh % Hh, b = bh / Hh;
  int lane = threadIdx.x;
  int col = lane & 15;
  int lanehi = (lane & 16) ? 8 : 0;
  int n0 = it * 16;

  const float* xrow = xin + ((size_t)b * Nn + n0 + col) * DIN;
  v8f acc[4] = {};

  for (int k0 = 0; k0 < DIN; k0 += 32) {
    // 1) issue all B-fragment loads (one clause of b128s)
    v16bf Bf[4];
#pragma unroll
    for (int t = 0; t < 4; ++t) {
      const __bf16* wrow = WT + ((size_t)h * 64 + t * 16 + col) * DIN + k0;
#pragma unroll
      for (int e = 0; e < 16; ++e)
        Bf[t][e] = wrow[kmap(e, lanehi)];
    }
    // 2) build A (f32 loads + cvt) while B loads are in flight
    v16bf A;
#pragma unroll
    for (int e = 0; e < 16; ++e)
      A[e] = (__bf16)xrow[k0 + kmap(e, lanehi)];
    // 3) consume
#pragma unroll
    for (int t = 0; t < 4; ++t)
      acc[t] = __builtin_amdgcn_wmma_f32_16x16x32_bf16(
          false, A, false, Bf[t], (short)0, acc[t], false, false);
  }

  // Transposed store: for fixed t, v-consecutive -> n-contiguous (b128 merge)
  size_t base = ((size_t)b * Hh + h) * Nn;             // row base in [B,H,N]
#pragma unroll
  for (int t = 0; t < 4; ++t) {
    __bf16* wout = WhT + (((size_t)b * Hh + h) * 64 + t * 16 + col) * Nn
                 + n0 + lanehi;
#pragma unroll
    for (int v = 0; v < 8; ++v)
      wout[v] = (__bf16)acc[t][v];
  }

  // f1/f2: reduce this row's 64 columns across the 16 lanes of each half
#pragma unroll
  for (int v = 0; v < 8; ++v) {
    float p1 = 0.0f, p2 = 0.0f;
#pragma unroll
    for (int t = 0; t < 4; ++t) {
      float wv = acc[t][v];
      p1 += wv * avec[h * 128 + t * 16 + col];
      p2 += wv * avec[h * 128 + 64 + t * 16 + col];
    }
#pragma unroll
    for (int m = 8; m >= 1; m >>= 1) {
      p1 += __shfl_xor(p1, m, 32);
      p2 += __shfl_xor(p2, m, 32);
    }
    if (col == 0) {            // lane 0 and lane 16 write distinct rows
      int n = n0 + v + lanehi;
      f1[base + n] = p1;
      f2[base + n] = p2;
    }
  }
}

// ---------------------------------------------------------------------------
// Softmax stats (flash pass 1): per (b,h,i) row-max and exp-sum.
// Never materializes the NxN logit tensor — logits are rank-1 + mask.
// ---------------------------------------------------------------------------
__global__ __launch_bounds__(256)
void gat_stats(const float* __restrict__ f1, const float* __restrict__ f2,
               const uint32_t* __restrict__ maskw,
               float* __restrict__ mrow, float* __restrict__ lrow)
{
  __shared__ float red[256];
  int i = blockIdx.x % Nn;
  int bh = blockIdx.x / Nn;
  int h = bh % Hh, b = bh / Hh;
  size_t base = ((size_t)b * Hh + h) * Nn;
  float fi = f1[base + i];
  const uint32_t* mws = maskw + ((size_t)b * Nn + i) * (Nn / 32);
  int t = threadIdx.x;

  float mx = -INFINITY;
  for (int j = t; j < Nn; j += 256) {
    float e = fi + f2[base + j];
    e = e > 0.0f ? e : GAT_ALPHA * e;
    bool on = (mws[j >> 5] >> (j & 31)) & 1u;
    e = on ? e : GAT_NEG_INF;
    mx = fmaxf(mx, e);
  }
  red[t] = mx; __syncthreads();
  for (int s = 128; s > 0; s >>= 1) {
    if (t < s) red[t] = fmaxf(red[t], red[t + s]);
    __syncthreads();
  }
  mx = red[0]; __syncthreads();

  float sum = 0.0f;
  for (int j = t; j < Nn; j += 256) {
    float e = fi + f2[base + j];
    e = e > 0.0f ? e : GAT_ALPHA * e;
    bool on = (mws[j >> 5] >> (j & 31)) & 1u;
    e = on ? e : GAT_NEG_INF;
    sum += __expf(e - mx);     // fully-masked row: e-mx==0 -> uniform, like jax
  }
  red[t] = sum; __syncthreads();
  for (int s = 128; s > 0; s >>= 1) {
    if (t < s) red[t] += red[t + s];
    __syncthreads();
  }
  if (t == 0) { mrow[base + i] = mx; lrow[base + i] = red[0]; }
}

// ---------------------------------------------------------------------------
// Flash pass 2: out[16,64] = softmax(e) @ Wh via bf16 WMMA; probabilities
// regenerated in-register per 16x32 A fragment; B fragments come from the
// f-major WhT so each lane issues contiguous 16-byte loads. All 8 b128
// loads issue first; the exp/leaky-ReLU A-build runs in their shadow.
// ---------------------------------------------------------------------------
__global__ __launch_bounds__(32)
void gat_attn(const __bf16* __restrict__ WhT, const float* __restrict__ f1,
              const float* __restrict__ f2, const float* __restrict__ mrow,
              const float* __restrict__ lrow, const uint32_t* __restrict__ maskw,
              float* __restrict__ out, int concat)
{
  const int ntiles = Nn / 16;
  int it = blockIdx.x % ntiles;
  int bh = blockIdx.x / ntiles;
  int h = bh % Hh, b = bh / Hh;
  int lane = threadIdx.x;
  int col = lane & 15;
  int lanehi = (lane & 16) ? 8 : 0;
  int i0 = it * 16;
  int i = i0 + col;                       // A-fragment row for this lane
  size_t base = ((size_t)b * Hh + h) * Nn;
  float fi = f1[base + i];
  float mi = mrow[base + i];
  float rinv = 1.0f / lrow[base + i];
  const uint32_t* mws = maskw + ((size_t)b * Nn + i) * (Nn / 32);
  const __bf16* whb = WhT + ((size_t)b * Hh + h) * 64 * Nn;   // [64][N]

  v8f acc[4] = {};
  for (int j0 = 0; j0 < Nn; j0 += 32) {
    // 1) issue all B-fragment loads for this j-step (8 x b128, one clause)
    v16bf Bf[4];
#pragma unroll
    for (int t = 0; t < 4; ++t) {
      const __bf16* wrow = whb + (size_t)(t * 16 + col) * Nn + j0;
#pragma unroll
      for (int e = 0; e < 16; ++e)
        Bf[t][e] = wrow[kmap(e, lanehi)];
    }
    if (j0 + 32 < Nn)                     // prefetch next tile into near cache
      __builtin_prefetch(whb + (size_t)(lane * 2) * Nn + j0 + 32, 0, 3);
    // 2) A-fragment probabilities (exp/VALU) overlap the outstanding loads
    uint32_t mword = mws[j0 >> 5];        // all 16 elems share this word
    v16bf A;
#pragma unroll
    for (int e = 0; e < 16; ++e) {
      int j = j0 + kmap(e, lanehi);
      float ev = fi + f2[base + j];
      ev = ev > 0.0f ? ev : GAT_ALPHA * ev;
      bool on = (mword >> (j & 31)) & 1u;
      ev = on ? ev : GAT_NEG_INF;
      A[e] = (__bf16)(__expf(ev - mi) * rinv);
    }
    // 3) consume: 4 back-to-back WMMAs
#pragma unroll
    for (int t = 0; t < 4; ++t)
      acc[t] = __builtin_amdgcn_wmma_f32_16x16x32_bf16(
          false, A, false, Bf[t], (short)0, acc[t], false, false);
  }

#pragma unroll
  for (int v = 0; v < 8; ++v) {
    int n = i0 + v + lanehi;
#pragma unroll
    for (int t = 0; t < 4; ++t) {
      int f = t * 16 + col;
      float val = acc[t][v];
      if (concat) {
        val = val > 0.0f ? val : (__expf(val) - 1.0f);   // ELU
        out[((size_t)b * Nn + n) * (Hh * 64) + h * 64 + f] = val;
      } else {
        out[(base + n) * 64 + f] = val;   // per-head, reduced later
      }
    }
  }
}

// ---------------------------------------------------------------------------
// Head mean + sigmoid + dot with hidden -> prob / logits
// ---------------------------------------------------------------------------
__global__ __launch_bounds__(64)
void gat_final(const float* __restrict__ o3, const float* __restrict__ hidden,
               float* __restrict__ out)
{
  __shared__ float red[64];
  int n = blockIdx.x % Nn;
  int b = blockIdx.x / Nn;
  int f = threadIdx.x;
  float s = 0.0f;
#pragma unroll
  for (int h = 0; h < Hh; ++h)
    s += o3[(((size_t)b * Hh + h) * Nn + n) * 64 + f];
  s = fsigmoid(s * (1.0f / Hh));
  red[f] = s * hidden[b * 64 + f];
  __syncthreads();
  for (int st = 32; st > 0; st >>= 1) {
    if (f < st) red[f] += red[f + st];
    __syncthreads();
  }
  if (f == 0) {
    float lg = red[0];
    out[(size_t)b * Nn + n] = fsigmoid(lg);                       // output 0
    out[(size_t)Bb * Nn + (size_t)Bb * 64 + (size_t)b * Nn + n] = lg; // output 2
  }
}

__global__ void copy_hidden(const float* __restrict__ hidden, float* __restrict__ out)
{
  int t = blockIdx.x * blockDim.x + threadIdx.x;
  if (t < Bb * 64) out[(size_t)Bb * Nn + t] = hidden[t];          // output 1
}

// ---------------------------------------------------------------------------
extern "C" void kernel_launch(void* const* d_in, const int* in_sizes, int n_in,
                              void* d_out, int out_size, void* d_ws, size_t ws_size,
                              hipStream_t stream) {
  (void)in_sizes; (void)n_in; (void)out_size; (void)ws_size;
  const int*   story  = (const int*)d_in[0];
  const int*   kb     = (const int*)d_in[1];
  const int*   cv     = (const int*)d_in[2];
  const float* hidden = (const float*)d_in[3];
  const float* dh     = (const float*)d_in[4];
  const float* adj    = (const float*)d_in[5];
  const float* emb    = (const float*)d_in[6];
  const float* W1     = (const float*)d_in[7];
  const float* a1     = (const float*)d_in[8];
  const float* W2     = (const float*)d_in[9];
  const float* a2     = (const float*)d_in[10];
  const float* W3     = (const float*)d_in[11];
  const float* a3     = (const float*)d_in[12];
  float* out = (float*)d_out;
  char*  ws  = (char*)d_ws;

  const size_t MB = 1024 * 1024;
  // Region plan (aliasing is safe across phases):
  //  [0,16M): xin [B,N,128] f32 (8M, dead after layer1) then o3 [B,H,N,64] f32
  //  [16,32M): x1 [B,N,256] f32   [32,48M): x2 [B,N,256] f32
  //  [48,50M): maskw               [50,58M): WhT bf16 [B,H,64,N]
  //  [58,59M): f1/f2/m/l (256K each)   [59,60M): WT bf16 [H,64,DIN]
  float*    xin  = (float*)(ws + 0);
  float*    o3   = (float*)(ws + 0);
  float*    x1   = (float*)(ws + 16 * MB);
  float*    x2   = (float*)(ws + 32 * MB);
  uint32_t* mskw = (uint32_t*)(ws + 48 * MB);
  __bf16*   WhT  = (__bf16*)(ws + 50 * MB);
  float*    f1   = (float*)(ws + 58 * MB);
  float*    f2   = (float*)(ws + 58 * MB + 256 * 1024);
  float*    mr   = (float*)(ws + 58 * MB + 512 * 1024);
  float*    lr   = (float*)(ws + 58 * MB + 768 * 1024);
  __bf16*   WT   = (__bf16*)(ws + 59 * MB);

  const int tiles = Bb * Hh * (Nn / 16);

  build_x<<<Bb * Nn, 128, 0, stream>>>(story, kb, cv, dh, emb, xin);
  build_mask<<<Bb * Nn, 32, 0, stream>>>(adj, kb, cv, mskw);

  // Layer 1 (Din=128, concat)
  conv_w<<<(Hh * 128 * 64 + 255) / 256, 256, 0, stream>>>(W1, WT, 128);
  gat_wh<128><<<tiles, 32, 0, stream>>>(xin, WT, a1, WhT, f1, f2);
  gat_stats<<<Bb * Hh * Nn, 256, 0, stream>>>(f1, f2, mskw, mr, lr);
  gat_attn<<<tiles, 32, 0, stream>>>(WhT, f1, f2, mr, lr, mskw, x1, 1);

  // Layer 2 (Din=256, concat)
  conv_w<<<(Hh * 256 * 64 + 255) / 256, 256, 0, stream>>>(W2, WT, 256);
  gat_wh<256><<<tiles, 32, 0, stream>>>(x1, WT, a2, WhT, f1, f2);
  gat_stats<<<Bb * Hh * Nn, 256, 0, stream>>>(f1, f2, mskw, mr, lr);
  gat_attn<<<tiles, 32, 0, stream>>>(WhT, f1, f2, mr, lr, mskw, x2, 1);

  // Layer 3 (Din=256, head-mean)
  conv_w<<<(Hh * 256 * 64 + 255) / 256, 256, 0, stream>>>(W3, WT, 256);
  gat_wh<256><<<tiles, 32, 0, stream>>>(x2, WT, a3, WhT, f1, f2);
  gat_stats<<<Bb * Hh * Nn, 256, 0, stream>>>(f1, f2, mskw, mr, lr);
  gat_attn<<<tiles, 32, 0, stream>>>(WhT, f1, f2, mr, lr, mskw, o3, 0);

  gat_final<<<Bb * Nn, 64, 0, stream>>>(o3, hidden, out);
  copy_hidden<<<(Bb * 64 + 255) / 256, 256, 0, stream>>>(hidden, out);
}
